// CausalVarlenSelfAttention_50663434223693
// MI455X (gfx1250) — matrix operand order repaired and checked by reading, not compiled, not run on hardware
//
#include <hip/hip_runtime.h>
#include <hip/hip_bf16.h>

typedef __attribute__((ext_vector_type(16))) _Float16 v16h;
typedef __attribute__((ext_vector_type(4)))  _Float16 v4h;
typedef __attribute__((ext_vector_type(8)))  float    v8f;
typedef __attribute__((ext_vector_type(4)))  unsigned int u32x4;
typedef __attribute__((ext_vector_type(8)))  int      i32x8;
typedef __attribute__((ext_vector_type(4)))  int      i32x4;

union Frag16 { v16h h; uint4 u[2]; };

static __device__ __forceinline__ v8f wmma_f16(const Frag16& a, const Frag16& b, v8f c) {
  // v_wmma_f32_16x16x32_f16: (neg_a, A, neg_b, B, c_mod, C, reuse_a, reuse_b)
  return __builtin_amdgcn_wmma_f32_16x16x32_f16(false, a.h, false, b.h, (short)0, c, false, false);
}

#if __has_builtin(__builtin_amdgcn_tensor_load_to_lds) && __has_builtin(__builtin_amdgcn_s_wait_tensorcnt)
#define USE_TDM 1
#else
#define USE_TDM 0
#endif

#if USE_TDM
// ---------------------------------------------------------------------------
// TDM: DMA a 64(rows) x 64(halves) 2-D tile (row stride = stride_elems halves)
// from global into LDS, padded to a 72-half row stride in LDS
// (pad_interval = 32 DWORDs -> one 128B row, pad_amount = 4 DWORDs = 16B).
// D# layout per CDNA5 ISA ch.8 (groups 0/1; groups 2/3 zero for 2-D tiles).
// ---------------------------------------------------------------------------
static __device__ __forceinline__ void tdm_load_tile64(
    unsigned int lds_byte_off, const _Float16* gsrc, unsigned int stride_elems) {
  unsigned long long ga = (unsigned long long)(const void*)gsrc;
  u32x4 g0;
  g0[0] = 1u;                                        // count=1 (valid user D#)
  g0[1] = lds_byte_off;                              // lds_addr
  g0[2] = (unsigned int)(ga & 0xFFFFFFFFu);          // global_addr[31:0]
  g0[3] = (unsigned int)((ga >> 32) & 0x1FFFFFFu)    // global_addr[56:32]
          | (2u << 30);                              // type = 2 ("image")
  i32x8 g1;
  g1[0] = (int)((1u << 16)      // data_size = 2 bytes
              | (1u << 20)      // pad_enable
              | (4u << 22)      // pad_interval: 32 DWORDs (one 128B row)
              | (3u << 25));    // pad_amount: 4 DWORDs (16B)
  g1[1] = (int)(64u << 16);     // tensor_dim0[15:0] = 64 (abar addr = 0)
  g1[2] = (int)(64u << 16);     // tensor_dim0 hi = 0 | tensor_dim1[15:0] = 64
  g1[3] = (int)(64u << 16);     // tensor_dim1 hi = 0 | tile_dim0 = 64
  g1[4] = (int)64;              // tile_dim1 = 64 | tile_dim2 = 0
  g1[5] = (int)stride_elems;    // tensor_dim0_stride[31:0]
  g1[6] = 0;                    // stride0 hi | stride1 lo (unused, 2-D)
  g1[7] = 0;
  i32x4 gz = {0, 0, 0, 0};
#if defined(__clang_major__) && (__clang_major__ >= 23)
  i32x8 gz8 = {0, 0, 0, 0, 0, 0, 0, 0};
  __builtin_amdgcn_tensor_load_to_lds(g0, g1, gz, gz, gz8, 0);
#else
  __builtin_amdgcn_tensor_load_to_lds(g0, g1, gz, gz, 0);
#endif
}
static __device__ __forceinline__ unsigned int lds_off_of(const void* p) {
  // LDS flat addresses carry the LDS byte offset in addr[31:0] (aperture in
  // the high half), so truncation yields the D# lds_addr.
  return (unsigned int)(unsigned long long)p;
}
#endif  // USE_TDM

// ---------------------------------------------------------------------------
// f32 -> f16 conversion (vectorized x4)
// ---------------------------------------------------------------------------
__global__ __launch_bounds__(256) void convert_f16_kernel(
    const float* __restrict__ src, _Float16* __restrict__ dst, int count4) {
  int i = blockIdx.x * blockDim.x + threadIdx.x;
  if (i < count4) {
    float4 f = ((const float4*)src)[i];
    v4h o = {(_Float16)f.x, (_Float16)f.y, (_Float16)f.z, (_Float16)f.w};
    ((v4h*)dst)[i] = o;
  }
}

// ---------------------------------------------------------------------------
// C[M,Ncols](f32) = A[M,K](f16) @ W[Ncols,K]^T (f16)   (A @ W^T, both K-major)
// Block tile 128x128, 8 waves, each wave 32(M) x 64(N): 2x4 WMMA C tiles.
// ---------------------------------------------------------------------------
__global__ __launch_bounds__(256) void gemm_f16_wmma(
    const _Float16* __restrict__ A, const _Float16* __restrict__ W,
    float* __restrict__ C, int M, int Ncols, int K) {
  const int lane = threadIdx.x & 31;
  const int wave = threadIdx.x >> 5;
  const int wm = wave >> 1;                 // 0..3
  const int wn = wave & 1;                  // 0..1
  const int row0 = blockIdx.x * 128 + wm * 32;
  const int col0 = blockIdx.y * 128 + wn * 64;
  const int l15 = lane & 15;
  const int hh  = lane >> 4;                // half-wave select

  v8f acc[2][4];
#pragma unroll
  for (int mi = 0; mi < 2; ++mi)
#pragma unroll
    for (int t = 0; t < 4; ++t) acc[mi][t] = (v8f){0.f,0.f,0.f,0.f,0.f,0.f,0.f,0.f};

  const _Float16* arow0 = A + (size_t)(row0 + l15) * K;
  const _Float16* arow1 = A + (size_t)(row0 + 16 + l15) * K;
  const _Float16* wrow[4];
#pragma unroll
  for (int t = 0; t < 4; ++t)
    wrow[t] = W + (size_t)(col0 + t * 16 + l15) * K + hh * 16;

  for (int k0 = 0; k0 < K; k0 += 32) {
    __builtin_prefetch(arow0 + k0 + 128, 0, 1);
    __builtin_prefetch(wrow[0] + k0 + 128, 0, 1);
    Frag16 a0, a1;
    a0.u[0] = *(const uint4*)(arow0 + k0 + hh * 16);
    a0.u[1] = *(const uint4*)(arow0 + k0 + 8 + hh * 16);
    a1.u[0] = *(const uint4*)(arow1 + k0 + hh * 16);
    a1.u[1] = *(const uint4*)(arow1 + k0 + 8 + hh * 16);
#pragma unroll
    for (int t = 0; t < 4; ++t) {
      Frag16 b;
      b.u[0] = *(const uint4*)(wrow[t] + k0);
      b.u[1] = *(const uint4*)(wrow[t] + k0 + 8);
      acc[0][t] = wmma_f16(a0, b, acc[0][t]);
      acc[1][t] = wmma_f16(a1, b, acc[1][t]);
    }
  }

#pragma unroll
  for (int mi = 0; mi < 2; ++mi)
#pragma unroll
    for (int t = 0; t < 4; ++t)
#pragma unroll
      for (int i = 0; i < 8; ++i) {
        int r = row0 + mi * 16 + i + 8 * hh;
        int c = col0 + t * 16 + l15;
        C[(size_t)r * Ncols + c] = acc[mi][t][i];
      }
}

// ---------------------------------------------------------------------------
// RoPE + RMSNorm on q,k ; V transposed to d-major (for K-contiguous P@V feed)
// One wave per (token, which in {q,k,v}, head). D=64, H=16.
// ---------------------------------------------------------------------------
__global__ __launch_bounds__(256) void rope_rms_split_kernel(
    const float* __restrict__ qkv, const int* __restrict__ pos,
    const float* __restrict__ cosc, const float* __restrict__ sinc,
    _Float16* __restrict__ qh, _Float16* __restrict__ kh,
    _Float16* __restrict__ vt, int Ntok, int S) {
  const int wid  = (int)((blockIdx.x * blockDim.x + threadIdx.x) >> 5);
  const int lane = threadIdx.x & 31;
  const int H = 16;
  if (wid >= Ntok * 3 * H) return;
  const int n = wid / (3 * H);
  const int r = wid - n * 3 * H;
  const int which = r >> 4;     // /H : 0=q 1=k 2=v
  const int h = r & 15;
  const float* src = qkv + (size_t)n * 3072 + which * 1024 + h * 64;
  float x1 = src[lane];
  float x2 = src[lane + 32];
  if (which == 2) {
    int b = n / S, s = n - b * S;
    _Float16* dst = vt + (size_t)(b * H + h) * 64 * S + s;
    dst[(size_t)lane * S]        = (_Float16)x1;
    dst[(size_t)(lane + 32) * S] = (_Float16)x2;
  } else {
    int p = pos[n];
    float c  = cosc[p * 32 + lane];
    float sn = sinc[p * 32 + lane];
    float o1 =  x1 * c  + x2 * sn;
    float o2 = -x1 * sn + x2 * c;
    float ss = o1 * o1 + o2 * o2;
#pragma unroll
    for (int m = 16; m; m >>= 1) ss += __shfl_xor(ss, m, 32);
    float inv = rsqrtf(ss * (1.0f / 64.0f) + 1e-6f);
    _Float16* dst = (which == 0 ? qh : kh) + (size_t)n * 1024 + h * 64;
    dst[lane]      = (_Float16)(o1 * inv);
    dst[lane + 32] = (_Float16)(o2 * inv);
  }
}

// ---------------------------------------------------------------------------
// gate[n,h] = sigmoid(x[n,:] . gate_w[h,:] + gate_b[h])  (tiny GEMV, f32)
// ---------------------------------------------------------------------------
__global__ __launch_bounds__(256) void gate_kernel(
    const float* __restrict__ x, const float* __restrict__ gw,
    const float* __restrict__ gb, float* __restrict__ gate, int Ntok) {
  int idx = blockIdx.x * blockDim.x + threadIdx.x;
  if (idx >= Ntok * 16) return;
  int n = idx >> 4, h = idx & 15;
  const float4* xr = (const float4*)(x + (size_t)n * 1024);
  const float4* wr = (const float4*)(gw + (size_t)h * 1024);
  float acc = 0.f;
  for (int i = 0; i < 256; ++i) {
    float4 a = xr[i], b = wr[i];
    acc += a.x * b.x + a.y * b.y + a.z * b.z + a.w * b.w;
  }
  acc += gb[h];
  gate[idx] = 1.0f / (1.0f + __expf(-acc));
}

// ---------------------------------------------------------------------------
// Flash-style causal attention, fused gate + 1/l epilogue.
// Grid: (S/128, B*H). Block 256 = 8 waves; each wave owns 16 q-rows.
// K/V chunks (64 keys x 64 dims) are staged into LDS by the Tensor Data
// Mover (double-buffered; wave0 moves K, wave1 moves V), padded to a
// 72-half row stride so ds_load_b128 fragment reads are aligned.
// ---------------------------------------------------------------------------
#define PSTR 72   // LDS row stride in halves (144B -> 16B aligned b128 reads)

__global__ __launch_bounds__(256) void attn_kernel(
    const _Float16* __restrict__ qh, const _Float16* __restrict__ kh,
    const _Float16* __restrict__ vt, const float* __restrict__ gate,
    _Float16* __restrict__ og, int S) {
#if USE_TDM
  __shared__ _Float16 kbuf[2][64 * PSTR];
  __shared__ _Float16 vbuf[2][64 * PSTR];
#endif
  __shared__ _Float16 plds[8 * 16 * PSTR];
  const int lane = threadIdx.x & 31;
  const int wave = threadIdx.x >> 5;
  const int l15 = lane & 15;
  const int hh  = lane >> 4;
  const int bh = blockIdx.y;
  const int b = bh >> 4;
  const int h = bh & 15;
  const int m0 = blockIdx.x * 128 + wave * 16;   // this wave's q-row base

  // Resident Q A-fragments (16 rows x 64 dims -> 2 frags of 16x32)
  const _Float16* qrow = qh + (size_t)(b * S + m0 + l15) * 1024 + h * 64;
  Frag16 qa[2];
  qa[0].u[0] = *(const uint4*)(qrow + hh * 16);
  qa[0].u[1] = *(const uint4*)(qrow + 8 + hh * 16);
  qa[1].u[0] = *(const uint4*)(qrow + 32 + hh * 16);
  qa[1].u[1] = *(const uint4*)(qrow + 40 + hh * 16);

  v8f o[4];
#pragma unroll
  for (int t = 0; t < 4; ++t) o[t] = (v8f){0.f,0.f,0.f,0.f,0.f,0.f,0.f,0.f};
  float mrow[8], lrow[8];
#pragma unroll
  for (int i = 0; i < 8; ++i) { mrow[i] = -3.0e38f; lrow[i] = 0.f; }

  _Float16* pbase = plds + wave * 16 * PSTR;
  const int nchunks = (blockIdx.x * 128 + 128) / 64;   // causal horizon
  const _Float16* kg = kh + (size_t)b * S * 1024 + h * 64;  // key rows, ld=1024
  const _Float16* vg = vt + (size_t)bh * 64 * S;            // d rows,  ld=S

#if USE_TDM
  if (wave == 0)      tdm_load_tile64(lds_off_of(kbuf[0]), kg, 1024u);
  else if (wave == 1) tdm_load_tile64(lds_off_of(vbuf[0]), vg, (unsigned)S);
#endif

  for (int c = 0; c < nchunks; ++c) {
    const int kb = c * 64;
#if USE_TDM
    const int p = c & 1;
    if (wave == 0) {
      if (c + 1 < nchunks) {
        tdm_load_tile64(lds_off_of(kbuf[p ^ 1]), kg + (size_t)(kb + 64) * 1024, 1024u);
        __builtin_amdgcn_s_wait_tensorcnt((short)1);
      } else {
        __builtin_amdgcn_s_wait_tensorcnt((short)0);
      }
    } else if (wave == 1) {
      if (c + 1 < nchunks) {
        tdm_load_tile64(lds_off_of(vbuf[p ^ 1]), vg + kb + 64, (unsigned)S);
        __builtin_amdgcn_s_wait_tensorcnt((short)1);
      } else {
        __builtin_amdgcn_s_wait_tensorcnt((short)0);
      }
    }
    __syncthreads();                 // K/V chunk c visible in LDS to all waves
    const _Float16* kc = kbuf[p];
    const _Float16* vc = vbuf[p];
#endif
    // ---- S = (Q K^T) * scale, causal-masked -------------------------------
    v8f sacc[4];
#pragma unroll
    for (int t = 0; t < 4; ++t) sacc[t] = (v8f){0.f,0.f,0.f,0.f,0.f,0.f,0.f,0.f};
#pragma unroll
    for (int t = 0; t < 4; ++t) {
#pragma unroll
      for (int j = 0; j < 2; ++j) {
        Frag16 bk;
#if USE_TDM
        const _Float16* ks = kc + (t * 16 + l15) * PSTR + j * 32 + hh * 16;
#else
        const _Float16* ks =
            kg + (size_t)(kb + t * 16 + l15) * 1024 + j * 32 + hh * 16;
#endif
        bk.u[0] = *(const uint4*)(ks);
        bk.u[1] = *(const uint4*)(ks + 8);
        sacc[t] = wmma_f16(qa[j], bk, sacc[t]);
      }
    }
#pragma unroll
    for (int t = 0; t < 4; ++t)
#pragma unroll
      for (int i = 0; i < 8; ++i) {
        int key = kb + t * 16 + l15;
        int qr  = m0 + i + 8 * hh;
        float vv = sacc[t][i] * 0.125f;          // D^-0.5
        sacc[t][i] = (key <= qr) ? vv : -3.0e38f;
      }
    // ---- online softmax ---------------------------------------------------
    float corr[8];
#pragma unroll
    for (int i = 0; i < 8; ++i) {
      float rm = fmaxf(fmaxf(sacc[0][i], sacc[1][i]),
                       fmaxf(sacc[2][i], sacc[3][i]));
#pragma unroll
      for (int msk = 8; msk; msk >>= 1) rm = fmaxf(rm, __shfl_xor(rm, msk, 32));
      float mn = fmaxf(mrow[i], rm);
      corr[i] = __expf(mrow[i] - mn);
      mrow[i] = mn;
    }
#pragma unroll
    for (int i = 0; i < 8; ++i) {
      float rs = 0.f;
#pragma unroll
      for (int t = 0; t < 4; ++t) {
        float pv = __expf(sacc[t][i] - mrow[i]);
        sacc[t][i] = pv;
        rs += pv;
      }
#pragma unroll
      for (int msk = 8; msk; msk >>= 1) rs += __shfl_xor(rs, msk, 32);
      lrow[i] = lrow[i] * corr[i] + rs;
#pragma unroll
      for (int t = 0; t < 4; ++t) o[t][i] *= corr[i];
    }
    // ---- stage P (f16) through this wave's private LDS region -------------
#pragma unroll
    for (int t = 0; t < 4; ++t)
#pragma unroll
      for (int i = 0; i < 8; ++i)
        pbase[(i + 8 * hh) * PSTR + t * 16 + l15] = (_Float16)sacc[t][i];
    // LDS ops are in-order within a wave; no barrier needed (private region)
#pragma unroll
    for (int j = 0; j < 2; ++j) {
      Frag16 pa;
      const _Float16* pr = pbase + l15 * PSTR + j * 32;
      pa.u[0] = *(const uint4*)(pr + hh * 16);
      pa.u[1] = *(const uint4*)(pr + 8 + hh * 16);
#pragma unroll
      for (int t = 0; t < 4; ++t) {
        Frag16 bv;
#if USE_TDM
        const _Float16* vs = vc + (t * 16 + l15) * PSTR + j * 32 + hh * 16;
#else
        const _Float16* vs =
            vg + (size_t)(t * 16 + l15) * S + kb + j * 32 + hh * 16;
#endif
        bv.u[0] = *(const uint4*)(vs);
        bv.u[1] = *(const uint4*)(vs + 8);
        o[t] = wmma_f16(pa, bv, o[t]);
      }
    }
#if USE_TDM
    __syncthreads();                 // release buffer p for the next TDM write
#endif
  }
  // ---- epilogue: 1/l, gate, f16 store ------------------------------------
#pragma unroll
  for (int t = 0; t < 4; ++t)
#pragma unroll
    for (int i = 0; i < 8; ++i) {
      int qr = m0 + i + 8 * hh;
      size_t ntok = (size_t)b * S + qr;
      float g = gate[ntok * 16 + h];
      float vv = o[t][i] / lrow[i] * g;
      og[ntok * 1024 + h * 64 + t * 16 + l15] = (_Float16)vv;
    }
}

// ---------------------------------------------------------------------------
extern "C" void kernel_launch(void* const* d_in, const int* in_sizes, int n_in,
                              void* d_out, int out_size, void* d_ws, size_t ws_size,
                              hipStream_t stream) {
  const float* x      = (const float*)d_in[0];
  const float* Wqkv   = (const float*)d_in[1];
  const float* Wo     = (const float*)d_in[2];
  const float* gate_w = (const float*)d_in[3];
  const float* gate_b = (const float*)d_in[4];
  const float* cosc   = (const float*)d_in[5];
  const float* sinc   = (const float*)d_in[6];
  const int*   pos    = (const int*)d_in[8];

  const int NH = 1024;                       // n_hidden
  const int H  = 16;
  const int N  = in_sizes[0] / NH;           // tokens (8192)
  const int B  = in_sizes[7] - 1;            // batches (8)
  const int S  = N / B;                      // seq len (1024)

  char* ws = (char*)d_ws;
  _Float16* xh    = (_Float16*)ws; ws += (size_t)N * NH * 2;
  _Float16* wqkvh = (_Float16*)ws; ws += (size_t)3 * NH * NH * 2;
  _Float16* woh   = (_Float16*)ws; ws += (size_t)NH * NH * 2;
  _Float16* qh    = (_Float16*)ws; ws += (size_t)N * NH * 2;
  _Float16* kh    = (_Float16*)ws; ws += (size_t)N * NH * 2;
  _Float16* vth   = (_Float16*)ws; ws += (size_t)N * NH * 2;
  float*    gate  = (float*)ws;    ws += (size_t)N * H * 4;
  float*    qkvf  = (float*)ws;    ws += (size_t)N * 3 * NH * 4;
  _Float16* attng = (_Float16*)qkvf;   // alias: qkvf dead once attention runs

  // 1) converts
  {
    int c4 = N * NH / 4;
    convert_f16_kernel<<<(c4 + 255) / 256, 256, 0, stream>>>(x, xh, c4);
    c4 = 3 * NH * NH / 4;
    convert_f16_kernel<<<(c4 + 255) / 256, 256, 0, stream>>>(Wqkv, wqkvh, c4);
    c4 = NH * NH / 4;
    convert_f16_kernel<<<(c4 + 255) / 256, 256, 0, stream>>>(Wo, woh, c4);
  }
  // 2) QKV projection: qkv[N,3072] = x @ Wqkv^T
  {
    dim3 grid(N / 128, (3 * NH) / 128);
    gemm_f16_wmma<<<grid, 256, 0, stream>>>(xh, wqkvh, qkvf, N, 3 * NH, NH);
  }
  // 3) RoPE + RMSNorm + split/transpose
  {
    int waves = 3 * N * H;
    rope_rms_split_kernel<<<(waves + 7) / 8, 256, 0, stream>>>(
        qkvf, pos, cosc, sinc, qh, kh, vth, N, S);
  }
  // 4) gate
  gate_kernel<<<(N * H + 255) / 256, 256, 0, stream>>>(x, gate_w, gate_b, gate, N);
  // 5) attention (writes attng, aliased over qkvf)
  {
    dim3 grid(S / 128, B * H);
    attn_kernel<<<grid, 256, 0, stream>>>(qh, kh, vth, gate, attng, S);
  }
  // 6) output projection: out = attng @ Wo^T  (f32 out)
  {
    dim3 grid(N / 128, NH / 128);
    gemm_f16_wmma<<<grid, 256, 0, stream>>>(attng, woh, (float*)d_out, N, NH, NH);
  }
}
